// QueST_80315888435764
// MI455X (gfx1250) — matrix operand order repaired and checked by reading, not compiled, not run on hardware
//
#include <hip/hip_runtime.h>
#include <hip/hip_bf16.h>

#define N_NODES 30000
#define N_EDGES 480000
#define KSUB 32
#define NPAIR 16384
#define DIN 1024
#define DH 512
#define DZ 128
#define DBN 8
#define DDB 16
#define DDEC0P 160   /* 144 padded to multiple of 32 */

typedef __attribute__((ext_vector_type(16))) __bf16 v16bf;
typedef __attribute__((ext_vector_type(8)))  float  v8f;

union FragAB { v16bf v; uint4 q[2]; unsigned short u[16]; };

__device__ __forceinline__ unsigned short f2bf(float f) {
    unsigned int u = __float_as_uint(f);
    unsigned int r = u + 0x7FFFu + ((u >> 16) & 1u);   // round-to-nearest-even
    return (unsigned short)(r >> 16);
}

// ---------------- elementwise helpers ----------------
__global__ void k_copy_f32x4(float4* __restrict__ dst, const float4* __restrict__ src, long long n4) {
    long long t = (long long)blockIdx.x * blockDim.x + threadIdx.x;
    if (t < n4) dst[t] = src[t];
}

// two f32 -> packed 2x bf16 per thread
__global__ void k_f32_to_bf16x2(unsigned int* __restrict__ dst, const float2* __restrict__ src, long long n2) {
    long long t = (long long)blockIdx.x * blockDim.x + threadIdx.x;
    if (t >= n2) return;
    float2 v = src[t];
    dst[t] = (unsigned int)f2bf(v.x) | ((unsigned int)f2bf(v.y) << 16);
}

__global__ void k_cvt_pad_w0(unsigned short* __restrict__ Wp, const float* __restrict__ W) {
    // [144,512] fp32 -> [160,512] bf16, pad rows 144..159 with zero
    int t = blockIdx.x * blockDim.x + threadIdx.x;
    if (t >= DDEC0P * DH) return;
    int r = t / DH, c = t % DH;
    Wp[t] = (r < (DZ + DDB)) ? f2bf(W[r * DH + c]) : (unsigned short)0;
}

// ---------------- GIN aggregation: agg[dst] += feat[src] ----------------
__global__ void k_scatter_add(float* __restrict__ agg, const float* __restrict__ feat,
                              const int* __restrict__ srcI, const int* __restrict__ dstI,
                              int E, int D) {
    long long t = (long long)blockIdx.x * blockDim.x + threadIdx.x;
    int D4 = D >> 2;
    long long total = (long long)E * D4;
    if (t >= total) return;
    int e  = (int)(t / D4);
    int f4 = (int)(t - (long long)e * D4);
    int s = srcI[e], d = dstI[e];
    float4 v = *(const float4*)(feat + (size_t)s * D + 4 * f4);
    float* o = agg + (size_t)d * D + 4 * f4;
    atomicAdd(o + 0, v.x);
    atomicAdd(o + 1, v.y);
    atomicAdd(o + 2, v.z);
    atomicAdd(o + 3, v.w);
}

// ---------------- WMMA bf16 GEMM: C[M,Nc] = A[M,K]@B[K,Nc] + bias ----------------
// Block: 128 threads = 4 waves. Block tile: 256 rows x 16 cols.
// Wave w handles row tiles (w + 4*t)*16, t=0..3 -> one B fragment feeds 4 WMMAs.
#define MT 4
__global__ __launch_bounds__(128)
void k_gemm_bf16(const unsigned short* __restrict__ A,   // [M,K] bf16 row-major, K%32==0
                 const unsigned short* __restrict__ B,   // [K,Nc] bf16 row-major
                 const float* __restrict__ bias,         // [Nc]
                 float* __restrict__ C,                  // [M,Nc] f32
                 int M, int K, int Nc) {
    // transposed B tile: sBt[col][k], stride 40 halves (80B, 16B-aligned rows)
    __shared__ __align__(16) unsigned short sBt[16 * 40];
    const int wave = threadIdx.x >> 5;
    const int lane = threadIdx.x & 31;
    const int n0   = blockIdx.y * 16;
    const int mLoc = lane & 15;
    const int hi   = lane >> 4;
    const int blockRow = blockIdx.x * 256;

    const unsigned short* Arow[MT];
    int rowT[MT];
#pragma unroll
    for (int t = 0; t < MT; ++t) {
        rowT[t] = blockRow + (wave + 4 * t) * 16;
        int ar = rowT[t] + mLoc;
        if (ar >= M) ar = M - 1;          // clamp loads; stores guarded
        Arow[t] = A + (size_t)ar * K;
    }

    const v8f vzero = {};
    v8f acc[MT];
#pragma unroll
    for (int t = 0; t < MT; ++t) acc[t] = vzero;

    const int kb = hi ? 16 : 0;
    const int a0 = hi ? 16 : 0;
    const int a1 = hi ? 24 : 8;

    for (int k0 = 0; k0 < K; k0 += 32) {
        // stage transposed B tile: thread -> (k = tid/4, cols 4*(tid%4)..+3)
        {
            int k  = threadIdx.x >> 2;
            int c4 = (threadIdx.x & 3) * 4;
            uint2 w = *(const uint2*)(B + (size_t)(k0 + k) * Nc + n0 + c4);
            const unsigned short* e = (const unsigned short*)&w;
            sBt[(c4 + 0) * 40 + k] = e[0];
            sBt[(c4 + 1) * 40 + k] = e[1];
            sBt[(c4 + 2) * 40 + k] = e[2];
            sBt[(c4 + 3) * 40 + k] = e[3];
        }
        __syncthreads();

        FragAB b;   // lane = column n0+mLoc; 16 contiguous K-halves -> 2x ds_load_b128
        b.q[0] = *(const uint4*)(sBt + mLoc * 40 + kb);
        b.q[1] = *(const uint4*)(sBt + mLoc * 40 + kb + 8);

#pragma unroll
        for (int t = 0; t < MT; ++t) {
            FragAB a;  // A in contiguous k order, same as B: lanes 0-15 K 0..15, lanes 16-31 K 16..31
            a.q[0] = *(const uint4*)(Arow[t] + k0 + a0);
            a.q[1] = *(const uint4*)(Arow[t] + k0 + a1);
            acc[t] = __builtin_amdgcn_wmma_f32_16x16x32_bf16(false, a.v, false, b.v,
                                                             (short)0, acc[t], false, false);
        }
        __syncthreads();
    }

    float bb = bias ? bias[n0 + mLoc] : 0.0f;
#pragma unroll
    for (int t = 0; t < MT; ++t) {
#pragma unroll
        for (int v = 0; v < 8; ++v) {
            int m   = hi ? (8 + v) : v;
            int row = rowT[t] + m;
            if (row < M) C[(size_t)row * Nc + n0 + mLoc] = acc[t][v] + bb;
        }
    }
}

// ---------------- LayerNorm (+ optional ReLU), in place, wave per row ----------------
__global__ __launch_bounds__(256)
void k_ln_relu(float* __restrict__ x, int rows, int D, int doRelu) {
    int wave = threadIdx.x >> 5, lane = threadIdx.x & 31;
    int row = blockIdx.x * 8 + wave;
    if (row >= rows) return;
    float* p = x + (size_t)row * D;
    float s = 0.f, s2 = 0.f;
    for (int f = lane; f < D; f += 32) { float v = p[f]; s += v; s2 += v * v; }
#pragma unroll
    for (int m = 16; m >= 1; m >>= 1) {
        s  += __shfl_xor(s,  m, 32);
        s2 += __shfl_xor(s2, m, 32);
    }
    float mean = s / D;
    float var  = s2 / D - mean * mean;
    float inv  = rsqrtf(var + 1e-5f);
    for (int f = lane; f < D; f += 32) {
        float v = (p[f] - mean) * inv;
        if (doRelu) v = fmaxf(v, 0.f);
        p[f] = v;
    }
}

// ---------------- subgraph mean pool (float4 over features) ----------------
__global__ void k_pool(float* __restrict__ out, const float* __restrict__ z,
                       const int* __restrict__ idx, int n) {
    int t = blockIdx.x * blockDim.x + threadIdx.x;          // over n * (DZ/4)
    if (t >= n * (DZ / 4)) return;
    int node = t / (DZ / 4), f4 = t - node * (DZ / 4);
    const int* ip = idx + node * KSUB;
    float4 s = make_float4(0.f, 0.f, 0.f, 0.f);
#pragma unroll 4
    for (int k = 0; k < KSUB; ++k) {
        float4 v = *(const float4*)(z + (size_t)ip[k] * DZ + 4 * f4);
        s.x += v.x; s.y += v.y; s.z += v.z; s.w += v.w;
    }
    const float r = 1.0f / KSUB;
    *(float4*)(out + (size_t)node * DZ + 4 * f4) = make_float4(s.x * r, s.y * r, s.z * r, s.w * r);
}

// ---------------- batch encoder MLP: [N,8]->relu[12]->[16] ----------------
__global__ void k_batch_enc(float* __restrict__ emb, const float* __restrict__ labels,
                            const float* __restrict__ W0, const float* __restrict__ b0,
                            const float* __restrict__ W1, const float* __restrict__ b1, int n) {
    int i = blockIdx.x * blockDim.x + threadIdx.x;
    if (i >= n) return;
    float in[DBN];
#pragma unroll
    for (int a = 0; a < DBN; ++a) in[a] = labels[i * DBN + a];
    float h[12];
#pragma unroll
    for (int j = 0; j < 12; ++j) {
        float s = b0[j];
#pragma unroll
        for (int a = 0; a < DBN; ++a) s += in[a] * W0[a * 12 + j];
        h[j] = fmaxf(s, 0.f);
    }
#pragma unroll
    for (int j = 0; j < DDB; ++j) {
        float s = b1[j];
#pragma unroll
        for (int a = 0; a < 12; ++a) s += h[a] * W1[a * DDB + j];
        emb[i * DDB + j] = s;
    }
}

// ---------------- batch discriminator: [N,128]->relu[64]->[8] ----------------
__global__ void k_bd(float* __restrict__ out, const float* __restrict__ zsub,
                     const float* __restrict__ W0, const float* __restrict__ b0,
                     const float* __restrict__ W1, const float* __restrict__ b1, int n) {
    int i = blockIdx.x * blockDim.x + threadIdx.x;
    if (i >= n) return;
    const float* z = zsub + (size_t)i * DZ;
    float h[64];
    for (int j = 0; j < 64; ++j) {
        float s = b0[j];
        for (int a = 0; a < DZ; ++a) s += z[a] * W0[a * 64 + j];
        h[j] = fmaxf(s, 0.f);
    }
    for (int j = 0; j < DBN; ++j) {
        float s = b1[j];
        for (int a = 0; a < 64; ++a) s += h[a] * W1[a * DBN + j];
        out[i * DBN + j] = s;
    }
}

// ---------------- bilinear contrastive logits: block per pair ----------------
__global__ __launch_bounds__(128)
void k_bilinear(float* __restrict__ lp, float* __restrict__ ln_,
                const float* __restrict__ zsub, const float* __restrict__ zshf,
                const float* __restrict__ W, const int* __restrict__ pos,
                const int* __restrict__ neg, int P) {
    __shared__ float red[128];
    int p = blockIdx.x;
    if (p >= P) return;
    int d = threadIdx.x;
    int ip = pos[p], in = neg[p];
    const float* zp  = zsub + (size_t)ip * DZ;
    const float* zsp = zshf + (size_t)ip * DZ;
    const float* zsn = zshf + (size_t)in * DZ;
    const float* Wd  = W + d * DZ;
    float tp = 0.f, tn = 0.f;
    for (int e = 0; e < DZ; ++e) { float w = Wd[e]; tp += w * zsp[e]; tn += w * zsn[e]; }
    float zd = zp[d];
    tp *= zd; tn *= zd;
    red[d] = tp; __syncthreads();
    for (int s = 64; s >= 1; s >>= 1) { if (d < s) red[d] += red[d + s]; __syncthreads(); }
    if (d == 0) lp[p] = red[0];
    __syncthreads();
    red[d] = tn; __syncthreads();
    for (int s = 64; s >= 1; s >>= 1) { if (d < s) red[d] += red[d + s]; __syncthreads(); }
    if (d == 0) ln_[p] = red[0];
}

// ---------------- concat z_node | batch_emb | zero-pad -> [N,160] ----------------
__global__ void k_concat(float* __restrict__ zb, const float* __restrict__ z,
                         const float* __restrict__ emb, int n) {
    int t = blockIdx.x * blockDim.x + threadIdx.x;
    if (t >= n * DDEC0P) return;
    int i = t / DDEC0P, c = t - i * DDEC0P;
    float v = 0.f;
    if (c < DZ) v = z[i * DZ + c];
    else if (c < DZ + DDB) v = emb[i * DDB + (c - DZ)];
    zb[t] = v;
}

// =====================================================================
extern "C" void kernel_launch(void* const* d_in, const int* in_sizes, int n_in,
                              void* d_out, int out_size, void* d_ws, size_t ws_size,
                              hipStream_t stream) {
    (void)in_sizes; (void)n_in; (void)out_size; (void)ws_size;
    const float* x        = (const float*)d_in[0];
    const float* x_shf    = (const float*)d_in[1];
    const float* blab     = (const float*)d_in[2];
    const float* enc_W0   = (const float*)d_in[3];
    const float* enc_b0   = (const float*)d_in[4];
    const float* enc_W1   = (const float*)d_in[5];
    const float* enc_b1   = (const float*)d_in[6];
    const float* dec_W0   = (const float*)d_in[7];
    const float* dec_b0   = (const float*)d_in[8];
    const float* dec_W1   = (const float*)d_in[9];
    const float* dec_b1   = (const float*)d_in[10];
    const float* be_W0    = (const float*)d_in[11];
    const float* be_b0    = (const float*)d_in[12];
    const float* be_W1    = (const float*)d_in[13];
    const float* be_b1    = (const float*)d_in[14];
    const float* bd_W0    = (const float*)d_in[15];
    const float* bd_b0    = (const float*)d_in[16];
    const float* bd_W1    = (const float*)d_in[17];
    const float* bd_b1    = (const float*)d_in[18];
    const float* bil_W    = (const float*)d_in[19];
    const int*   edge     = (const int*)d_in[20];
    const int*   sub_idx  = (const int*)d_in[21];
    const int*   pos_ind  = (const int*)d_in[22];
    const int*   neg_ind  = (const int*)d_in[23];
    const int* esrc = edge;
    const int* edst = edge + N_EDGES;

    // ---- output regions (float32, concatenated in return order) ----
    float* out = (float*)d_out;
    float* o_zsub   = out;                                  // [N,Z]
    float* o_zsubS  = o_zsub  + (size_t)N_NODES * DZ;       // [N,Z]
    float* o_recon  = o_zsubS + (size_t)N_NODES * DZ;       // [N,IN]
    float* o_lpos   = o_recon + (size_t)N_NODES * DIN;      // [P]
    float* o_lneg   = o_lpos  + NPAIR;                      // [P]
    float* o_lbatch = o_lneg  + NPAIR;                      // [N,BN]

    // ---- workspace carve-out ----
    char* ws = (char*)d_ws;
    size_t off = 0;
    auto carve = [&](size_t bytes) -> void* {
        void* p = ws + off;
        off = (off + bytes + 255) & ~(size_t)255;
        return p;
    };
    float*          bufA   = (float*)carve((size_t)N_NODES * DIN * 4);   // x+agg scratch (widest)
    unsigned short* bufAb  = (unsigned short*)carve((size_t)N_NODES * DIN * 2);
    float*          bufB   = (float*)carve((size_t)N_NODES * DH * 4);    // gemm out / h
    float*          z_node = (float*)carve((size_t)N_NODES * DZ * 4);
    float*          bemb   = (float*)carve((size_t)N_NODES * DDB * 4);
    float*          zb     = (float*)carve((size_t)N_NODES * DDEC0P * 4);
    unsigned short* encW0b = (unsigned short*)carve((size_t)DIN * DH * 2);
    unsigned short* encW1b = (unsigned short*)carve((size_t)DH * DZ * 2);
    unsigned short* decW0p = (unsigned short*)carve((size_t)DDEC0P * DH * 2);
    unsigned short* decW1b = (unsigned short*)carve((size_t)DH * DIN * 2);

    const int TB = 256;
    auto gl = [](long long n, int tb) { return (unsigned)((n + tb - 1) / tb); };
    auto cvt = [&](unsigned short* dst, const float* src, long long n) {
        k_f32_to_bf16x2<<<gl(n / 2, TB), TB, 0, stream>>>((unsigned int*)dst, (const float2*)src, n / 2);
    };
    auto cpy = [&](float* dst, const float* src, long long n) {
        k_copy_f32x4<<<gl(n / 4, TB), TB, 0, stream>>>((float4*)dst, (const float4*)src, n / 4);
    };

    // ---- weight conversion to bf16 ----
    cvt(encW0b, enc_W0, (long long)DIN * DH);
    cvt(encW1b, enc_W1, (long long)DH * DZ);
    k_cvt_pad_w0<<<gl((long long)DDEC0P * DH, TB), TB, 0, stream>>>(decW0p, dec_W0);
    cvt(decW1b, dec_W1, (long long)DH * DIN);

    // ---- batch encoder ----
    k_batch_enc<<<gl(N_NODES, TB), TB, 0, stream>>>(bemb, blab, be_W0, be_b0, be_W1, be_b1, N_NODES);

    const unsigned gmx = (N_NODES + 255) / 256;   // gemm grid.x (256 rows per block)

    // ---- encoder pass helper ----
    auto encode = [&](const float* xin, float* zout) {
        long long n0 = (long long)N_NODES * DIN;
        cpy(bufA, xin, n0);
        k_scatter_add<<<gl((long long)N_EDGES * (DIN / 4), TB), TB, 0, stream>>>(bufA, xin, esrc, edst, N_EDGES, DIN);
        cvt(bufAb, bufA, n0);
        k_gemm_bf16<<<dim3(gmx, DH / 16), 128, 0, stream>>>(bufAb, encW0b, enc_b0, bufB, N_NODES, DIN, DH);
        k_ln_relu<<<gl(N_NODES, 8), 256, 0, stream>>>(bufB, N_NODES, DH, 1);
        long long n1 = (long long)N_NODES * DH;
        cpy(bufA, bufB, n1);
        k_scatter_add<<<gl((long long)N_EDGES * (DH / 4), TB), TB, 0, stream>>>(bufA, bufB, esrc, edst, N_EDGES, DH);
        cvt(bufAb, bufA, n1);
        k_gemm_bf16<<<dim3(gmx, DZ / 16), 128, 0, stream>>>(bufAb, encW1b, enc_b1, zout, N_NODES, DH, DZ);
        k_ln_relu<<<gl(N_NODES, 8), 256, 0, stream>>>(zout, N_NODES, DZ, 1);
    };

    // shuffled first (z_node buffer reused afterwards for the real pass)
    encode(x_shf, z_node);
    k_pool<<<gl((long long)N_NODES * (DZ / 4), TB), TB, 0, stream>>>(o_zsubS, z_node, sub_idx, N_NODES);
    encode(x, z_node);
    k_pool<<<gl((long long)N_NODES * (DZ / 4), TB), TB, 0, stream>>>(o_zsub, z_node, sub_idx, N_NODES);

    // ---- batch discriminator + bilinear contrastive logits ----
    k_bd<<<gl(N_NODES, TB), TB, 0, stream>>>(o_lbatch, o_zsub, bd_W0, bd_b0, bd_W1, bd_b1, N_NODES);
    k_bilinear<<<NPAIR, 128, 0, stream>>>(o_lpos, o_lneg, o_zsub, o_zsubS, bil_W, pos_ind, neg_ind, NPAIR);

    // ---- decoder ----
    k_concat<<<gl((long long)N_NODES * DDEC0P, TB), TB, 0, stream>>>(zb, z_node, bemb, N_NODES);
    long long nd = (long long)N_NODES * DDEC0P;
    cpy(bufA, zb, nd);
    k_scatter_add<<<gl((long long)N_EDGES * (DDEC0P / 4), TB), TB, 0, stream>>>(bufA, zb, esrc, edst, N_EDGES, DDEC0P);
    cvt(bufAb, bufA, nd);
    k_gemm_bf16<<<dim3(gmx, DH / 16), 128, 0, stream>>>(bufAb, decW0p, dec_b0, bufB, N_NODES, DDEC0P, DH);
    k_ln_relu<<<gl(N_NODES, 8), 256, 0, stream>>>(bufB, N_NODES, DH, 1);
    long long n1 = (long long)N_NODES * DH;
    cpy(bufA, bufB, n1);
    k_scatter_add<<<gl((long long)N_EDGES * (DH / 4), TB), TB, 0, stream>>>(bufA, bufB, esrc, edst, N_EDGES, DH);
    cvt(bufAb, bufA, n1);
    // final layer: no norm, no act -> straight into recon output
    k_gemm_bf16<<<dim3(gmx, DIN / 16), 128, 0, stream>>>(bufAb, decW1b, dec_b1, o_recon, N_NODES, DH, DIN);
}
